// GroupedQueryAttention_15891378995337
// MI455X (gfx1250) — matrix operand order repaired and checked by reading, not compiled, not run on hardware
//
#include <hip/hip_runtime.h>
#include <hip/hip_bf16.h>

typedef __attribute__((ext_vector_type(16))) _Float16 v16h;
typedef __attribute__((ext_vector_type(8)))  _Float16 v8h;
typedef __attribute__((ext_vector_type(8)))  float    v8f;

// exact parameter type of the async-copy builtin (from hipcc diagnostic)
typedef int v4i __attribute__((vector_size(16)));
typedef __attribute__((address_space(1))) v4i gv4i;  // global
typedef __attribute__((address_space(3))) v4i lv4i;  // LDS

#define BB   4
#define TT   2048
#define DD   1024
#define HH   16
#define HD   64
#define MM   (BB*TT)     /* 8192 */
#define NQKV (3*DD)      /* 3072 */

// ---------------------------------------------------------------------------
// async global->LDS 16B copy (CDNA5 GLOBAL_LOAD_ASYNC_TO_LDS_B128, ASYNCcnt)
// ---------------------------------------------------------------------------
#if defined(__gfx1250__) && __has_builtin(__builtin_amdgcn_global_load_async_to_lds_b128)
#define ASYNC_CP16(ldst, gsrc)                                                  \
  __builtin_amdgcn_global_load_async_to_lds_b128(                               \
      (gv4i*)(gsrc), (lv4i*)(ldst), 0, 0)
#if __has_builtin(__builtin_amdgcn_s_wait_asynccnt)
#define ASYNC_WAIT() __builtin_amdgcn_s_wait_asynccnt(0)
#else
#define ASYNC_WAIT() asm volatile("s_wait_asynccnt 0x0" ::: "memory")
#endif
#else
// fallback: register-staged copy
#define ASYNC_CP16(ldst, gsrc)                                                  \
  (*reinterpret_cast<v8h*>(ldst) = *reinterpret_cast<const v8h*>(gsrc))
#define ASYNC_WAIT() ((void)0)
#endif

// ---------------------------------------------------------------------------
// helpers
// ---------------------------------------------------------------------------
static __device__ __forceinline__ v16h wmma_ld_a(const _Float16* row, int hi) {
  // A 16x32 f16 fragment: lane<16 holds K{0..15}, lane>=16 K{16..31}
  v16h a;
  *reinterpret_cast<v8h*>(&a)       = *reinterpret_cast<const v8h*>(row + hi * 16);
  *(reinterpret_cast<v8h*>(&a) + 1) = *reinterpret_cast<const v8h*>(row + hi * 16 + 8);
  return a;
}

static __device__ __forceinline__ v8f wmma_f16(v16h a, v16h b, v8f c) {
  return __builtin_amdgcn_wmma_f32_16x16x32_f16(false, a, false, b, (short)0, c,
                                                false, false);
}

// ---------------------------------------------------------------------------
// stage 0: conversions
// ---------------------------------------------------------------------------
__global__ __launch_bounds__(256) void cvt_f32_f16(const float* __restrict__ src,
                                                   _Float16* __restrict__ dst, int n) {
  int i = blockIdx.x * 256 + threadIdx.x;
  if (i < n) dst[i] = (_Float16)src[i];
}

// dst[n*K + k] = src[k*N + n]  (weights stored transposed for contiguous B frags)
__global__ __launch_bounds__(256) void cvt_f32_f16_t(const float* __restrict__ src,
                                                     _Float16* __restrict__ dst,
                                                     int K, int N) {
  int i = blockIdx.x * 256 + threadIdx.x;
  if (i < K * N) {
    int k = i / N, n = i % N;
    dst[(size_t)n * K + k] = (_Float16)src[i];
  }
}

// ---------------------------------------------------------------------------
// stage 1: fused QKV GEMM + bias + RoPE + head-split / transpose
//   Xh [M][D] f16,  WT [3D][D] f16 (transposed)
//   Qh [b*h][t][64], Kh [b*h][t][64], VT [b*h][64][T]
// ---------------------------------------------------------------------------
__global__ __launch_bounds__(256) void qkv_rope_kernel(
    const _Float16* __restrict__ Xh, const _Float16* __restrict__ WT,
    const float* __restrict__ bias, _Float16* __restrict__ Qh,
    _Float16* __restrict__ Kh, _Float16* __restrict__ VT) {
  __shared__ _Float16 As[128 * 32];
  __shared__ _Float16 Bs[128 * 32];

  const int tid  = threadIdx.x;
  const int wave = tid >> 5, lane = tid & 31;
  const int r = lane & 15, hi = lane >> 4;
  const int wm = wave & 3, wn = wave >> 2;     // 4 x 2 wave grid, wave = 32x64
  const int m0 = blockIdx.x * 128, n0 = blockIdx.y * 128;

  v8f acc[2][4] = {};

  for (int kt = 0; kt < 32; ++kt) {
    const int k0 = kt * 32;
#pragma unroll
    for (int c = tid; c < 512; c += 256) {
      int row = c >> 2, kc = (c & 3) << 3;
      ASYNC_CP16(&As[row * 32 + kc], &Xh[(size_t)(m0 + row) * DD + k0 + kc]);
      ASYNC_CP16(&Bs[row * 32 + kc], &WT[(size_t)(n0 + row) * DD + k0 + kc]);
    }
    if (kt + 1 < 32) {  // emits global_prefetch_b8 for next tile
      __builtin_prefetch(&Xh[(size_t)(m0 + (tid >> 1)) * DD + k0 + 32], 0, 1);
      __builtin_prefetch(&WT[(size_t)(n0 + (tid >> 1)) * DD + k0 + 32], 0, 1);
    }
    ASYNC_WAIT();
    __syncthreads();

    v16h af[2], bf[4];
#pragma unroll
    for (int ms = 0; ms < 2; ++ms)
      af[ms] = wmma_ld_a(&As[(wm * 32 + ms * 16 + r) * 32], hi);
#pragma unroll
    for (int ns = 0; ns < 4; ++ns)
      bf[ns] = *reinterpret_cast<const v16h*>(
          &Bs[(wn * 64 + ns * 16 + r) * 32 + hi * 16]);
#pragma unroll
    for (int ms = 0; ms < 2; ++ms)
#pragma unroll
      for (int ns = 0; ns < 4; ++ns)
        acc[ms][ns] = wmma_f16(af[ms], bf[ns], acc[ms][ns]);
    __syncthreads();
  }

  // epilogue: bias, RoPE (pairs are adjacent lanes in the C fragment), scatter
#pragma unroll
  for (int ms = 0; ms < 2; ++ms) {
#pragma unroll
    for (int ns = 0; ns < 4; ++ns) {
      const int n   = n0 + wn * 64 + ns * 16 + r;
      const float bval = bias[n];
      const int sec = n >> 10;          // 0=Q 1=K 2=V (uniform per fragment)
      const int nin = n & 1023;
      const int h = nin >> 6, ch = nin & 63;
      const float theta = __powf(1000.0f, -(2.0f * (float)((ch >> 1) + 1)) / 64.0f);
#pragma unroll
      for (int g = 0; g < 8; ++g) {
        const int m = m0 + wm * 32 + ms * 16 + hi * 8 + g;
        const int b = m >> 11, t = m & 2047;
        const int bh = b * HH + h;
        float val = acc[ms][ns][g] + bval;
        float partner = __shfl_xor(val, 1, 32);   // channel ch^1, same row
        if (sec < 2) {
          float ang = (float)(t + 1) * theta;
          float sn = __sinf(ang), cs = __cosf(ang);
          float rot = (ch & 1) ? (val * cs + partner * sn)
                               : (val * cs - partner * sn);
          if (sec == 0)
            Qh[((size_t)bh * TT + t) * HD + ch] = (_Float16)rot;
          else
            Kh[((size_t)bh * TT + t) * HD + ch] = (_Float16)rot;
        } else {
          VT[((size_t)bh * HD + ch) * TT + t] = (_Float16)val;
        }
      }
    }
  }
}

// ---------------------------------------------------------------------------
// stage 2: attention. One WG = one (b,h) x 32-query tile.
// Full 32x2048 f32 score rows in LDS (262KB of the 320KB WGP LDS)
// -> exact two-pass softmax, everything else WMMA.
// ---------------------------------------------------------------------------
#define ATTN_SMEM (32 * 2048 * 4 + 32 * 64 * 2 + 256 * 4 + 64 * 4)

__global__ __launch_bounds__(256) void attn_kernel(
    const _Float16* __restrict__ Qh, const _Float16* __restrict__ Kh,
    const _Float16* __restrict__ VT, _Float16* __restrict__ Oh) {
  extern __shared__ char smem[];
  float*    S      = reinterpret_cast<float*>(smem);                  // [32][2048]
  _Float16* Qs     = reinterpret_cast<_Float16*>(smem + 262144);      // [32][64]
  float*    red    = reinterpret_cast<float*>(smem + 262144 + 4096);  // [32][8]
  float*    rowmax = red + 256;                                       // [32]
  float*    rowsum = rowmax + 32;                                     // [32]

  const int tid  = threadIdx.x;
  const int wave = tid >> 5, lane = tid & 31;
  const int r = lane & 15, hi = lane >> 4;
  const int bh = blockIdx.x >> 6;
  const int q0 = (blockIdx.x & 63) * 32;
  const float scale = 0.04419417382415922f;  // 1/sqrt(512) = 1/sqrt(group_dim)

  // stage Q tile (32x64 f16) via async copy
  {
    int row = tid >> 3, col = (tid & 7) * 8;
    ASYNC_CP16(&Qs[row * 64 + col],
               &Qh[((size_t)bh * TT + q0 + row) * HD + col]);
  }
  ASYNC_WAIT();
  __syncthreads();

  // ---- phase 1: S = scale * Q K^T  (each wave strides over 16-key tiles)
  for (int nt = wave; nt < TT / 16; nt += 8) {
    const int n0k = nt * 16;
    v16h bf[2];
#pragma unroll
    for (int ks = 0; ks < 2; ++ks)
      bf[ks] = *reinterpret_cast<const v16h*>(
          &Kh[((size_t)bh * TT + n0k + r) * HD + ks * 32 + hi * 16]);
#pragma unroll
    for (int ms = 0; ms < 2; ++ms) {
      v8f acc = {};
#pragma unroll
      for (int ks = 0; ks < 2; ++ks) {
        v16h af = wmma_ld_a(&Qs[(ms * 16 + r) * 64 + ks * 32], hi);
        acc = wmma_f16(af, bf[ks], acc);
      }
#pragma unroll
      for (int g = 0; g < 8; ++g)
        S[(ms * 16 + hi * 8 + g) * 2048 + n0k + r] = acc[g] * scale;
    }
  }
  __syncthreads();

  // ---- phase 2: softmax over each of the 32 rows (8 threads / row)
  {
    const int row = tid >> 3, j = tid & 7;
    float* srow = &S[row * 2048 + j * 256];
    float mx = -3.0e30f;
    for (int i = 0; i < 256; ++i) mx = fmaxf(mx, srow[i]);
    red[row * 8 + j] = mx;
    __syncthreads();
    if (j == 0) {
      float m2 = red[row * 8];
      for (int i = 1; i < 8; ++i) m2 = fmaxf(m2, red[row * 8 + i]);
      rowmax[row] = m2;
    }
    __syncthreads();
    const float rm = rowmax[row];
    float sum = 0.0f;
    for (int i = 0; i < 256; ++i) {
      float e = __expf(srow[i] - rm);
      srow[i] = e;
      sum += e;
    }
    red[row * 8 + j] = sum;
    __syncthreads();
    if (j == 0) {
      float s2 = 0.0f;
      for (int i = 0; i < 8; ++i) s2 += red[row * 8 + i];
      rowsum[row] = s2;
    }
    __syncthreads();
  }

  // ---- phase 3: O = P V (wave = 16 queries x 16 channels; 64 k-steps)
  {
    const int msub = wave >> 2, nsub = wave & 3;
    v8f acc = {};
    for (int kt = 0; kt < 64; ++kt) {
      const int k0 = kt * 32;
      const float* srow = &S[(msub * 16 + r) * 2048 + k0];
      v16h af;
#pragma unroll
      for (int i = 0; i < 8; ++i) {
        af[i]     = (_Float16)srow[hi * 16 + i];
        af[8 + i] = (_Float16)srow[hi * 16 + 8 + i];
      }
      v16h bf = *reinterpret_cast<const v16h*>(
          &VT[((size_t)bh * HD + nsub * 16 + r) * TT + k0 + hi * 16]);
      acc = wmma_f16(af, bf, acc);
    }
    const int b = bh >> 4, h = bh & 15;
    const int ch = nsub * 16 + r;
#pragma unroll
    for (int g = 0; g < 8; ++g) {
      const int ml = msub * 16 + hi * 8 + g;
      const int t = q0 + ml;
      float o = acc[g] / rowsum[ml];
      Oh[((size_t)(b * TT + t)) * DD + h * HD + ch] = (_Float16)o;
    }
  }
}

// ---------------------------------------------------------------------------
// stage 3: output projection  out = Oh @ Wproj + b  (f32 out)
// ---------------------------------------------------------------------------
__global__ __launch_bounds__(256) void proj_kernel(
    const _Float16* __restrict__ Oh, const _Float16* __restrict__ WT,
    const float* __restrict__ bias, float* __restrict__ out) {
  __shared__ _Float16 As[128 * 32];
  __shared__ _Float16 Bs[128 * 32];

  const int tid  = threadIdx.x;
  const int wave = tid >> 5, lane = tid & 31;
  const int r = lane & 15, hi = lane >> 4;
  const int wm = wave & 3, wn = wave >> 2;
  const int m0 = blockIdx.x * 128, n0 = blockIdx.y * 128;

  v8f acc[2][4] = {};

  for (int kt = 0; kt < 32; ++kt) {
    const int k0 = kt * 32;
#pragma unroll
    for (int c = tid; c < 512; c += 256) {
      int row = c >> 2, kc = (c & 3) << 3;
      ASYNC_CP16(&As[row * 32 + kc], &Oh[(size_t)(m0 + row) * DD + k0 + kc]);
      ASYNC_CP16(&Bs[row * 32 + kc], &WT[(size_t)(n0 + row) * DD + k0 + kc]);
    }
    if (kt + 1 < 32) {
      __builtin_prefetch(&Oh[(size_t)(m0 + (tid >> 1)) * DD + k0 + 32], 0, 1);
      __builtin_prefetch(&WT[(size_t)(n0 + (tid >> 1)) * DD + k0 + 32], 0, 1);
    }
    ASYNC_WAIT();
    __syncthreads();

    v16h af[2], bf[4];
#pragma unroll
    for (int ms = 0; ms < 2; ++ms)
      af[ms] = wmma_ld_a(&As[(wm * 32 + ms * 16 + r) * 32], hi);
#pragma unroll
    for (int ns = 0; ns < 4; ++ns)
      bf[ns] = *reinterpret_cast<const v16h*>(
          &Bs[(wn * 64 + ns * 16 + r) * 32 + hi * 16]);
#pragma unroll
    for (int ms = 0; ms < 2; ++ms)
#pragma unroll
      for (int ns = 0; ns < 4; ++ns)
        acc[ms][ns] = wmma_f16(af[ms], bf[ns], acc[ms][ns]);
    __syncthreads();
  }

#pragma unroll
  for (int ms = 0; ms < 2; ++ms)
#pragma unroll
    for (int ns = 0; ns < 4; ++ns) {
      const int n = n0 + wn * 64 + ns * 16 + r;
      const float bval = bias[n];
#pragma unroll
      for (int g = 0; g < 8; ++g) {
        const int m = m0 + wm * 32 + ms * 16 + hi * 8 + g;
        out[(size_t)m * DD + n] = acc[ms][ns][g] + bval;
      }
    }
}

// ---------------------------------------------------------------------------
// launch
// ---------------------------------------------------------------------------
extern "C" void kernel_launch(void* const* d_in, const int* in_sizes, int n_in,
                              void* d_out, int out_size, void* d_ws, size_t ws_size,
                              hipStream_t stream) {
  const float* x      = (const float*)d_in[0];
  const float* W_attn = (const float*)d_in[1];
  const float* b_attn = (const float*)d_in[2];
  const float* W_proj = (const float*)d_in[3];
  const float* b_proj = (const float*)d_in[4];
  float* out = (float*)d_out;

  char* ws = (char*)d_ws;
  size_t off = 0;
  _Float16* Xh  = (_Float16*)(ws + off); off += (size_t)MM * DD * 2;     // 16 MB
  _Float16* WaT = (_Float16*)(ws + off); off += (size_t)NQKV * DD * 2;   //  6 MB
  _Float16* WpT = (_Float16*)(ws + off); off += (size_t)DD * DD * 2;     //  2 MB
  _Float16* Qh  = (_Float16*)(ws + off); off += (size_t)MM * DD * 2;     // 16 MB
  _Float16* Kh  = (_Float16*)(ws + off); off += (size_t)MM * DD * 2;     // 16 MB
  _Float16* VT  = (_Float16*)(ws + off); off += (size_t)MM * DD * 2;     // 16 MB
  _Float16* Oh  = (_Float16*)(ws + off); off += (size_t)MM * DD * 2;     // 16 MB

  // stage 0: precision conversion (+ weight transpose for contiguous B frags)
  cvt_f32_f16<<<(MM * DD + 255) / 256, 256, 0, stream>>>(x, Xh, MM * DD);
  cvt_f32_f16_t<<<(DD * NQKV + 255) / 256, 256, 0, stream>>>(W_attn, WaT, DD, NQKV);
  cvt_f32_f16_t<<<(DD * DD + 255) / 256, 256, 0, stream>>>(W_proj, WpT, DD, DD);

  // stage 1: QKV GEMM + bias + RoPE + layout
  qkv_rope_kernel<<<dim3(MM / 128, NQKV / 128), 256, 0, stream>>>(
      Xh, WaT, b_attn, Qh, Kh, VT);

  // stage 2: attention (big-LDS exact softmax)
  (void)hipFuncSetAttribute((const void*)attn_kernel,
                            hipFuncAttributeMaxDynamicSharedMemorySize, ATTN_SMEM);
  attn_kernel<<<dim3(BB * HH * (TT / 32)), 256, ATTN_SMEM, stream>>>(Qh, Kh, VT, Oh);

  // stage 3: output projection
  proj_kernel<<<dim3(MM / 128, DD / 128), 256, 0, stream>>>(Oh, WpT, b_proj, out);
}
